// Glm4MoeMoEExpertParallel_9981503995985
// MI455X (gfx1250) — matrix operand order repaired and checked by reading, not compiled, not run on hardware
//
#include <hip/hip_runtime.h>
#include <hip/hip_bf16.h>

// ---------------- problem constants (match reference) ----------------
constexpr int Bb = 2, Ss = 4096, Hh = 1024;
constexpr int Tt = Bb * Ss;          // 8192 tokens
constexpr int Ii = 512;              // intermediate
constexpr int Ee = 8;                // routed experts
constexpr int NEXP = Ee + 1;         // + shared pseudo-expert
constexpr float ROUTED_SCALE = 1.0f;
constexpr int MT = 32;               // tokens per block (2 WMMA M-subtiles)

typedef __attribute__((ext_vector_type(16))) __bf16 v16bf;
typedef __attribute__((ext_vector_type(8)))  float  v8f;

union BFrag { v16bf v; uint4 q[2]; };

__device__ __forceinline__ unsigned short f2bf(float f) {
  unsigned u = __float_as_uint(f);
  unsigned r = u + 0x7FFFu + ((u >> 16) & 1u);   // round-to-nearest-even
  return (unsigned short)(r >> 16);
}

__device__ __forceinline__ float fast_sigmoid(float x) {
  return __builtin_amdgcn_rcpf(1.0f + __expf(-x));   // v_rcp_f32, no div_scale chain
}

// 32x16 bf16 B fragment: lane's 16 values contiguous (pre-swizzled)
__device__ __forceinline__ v16bf load_bfrag(const unsigned short* p) {
  BFrag f;
  const uint4* q = (const uint4*)p;
  f.q[0] = q[0];
  f.q[1] = q[1];
  return f.v;
}

// 16x32 bf16 A fragment from LDS row-major: lane l -> row l&15, K runs [kb,kb+8) and [kb+16,kb+24)
__device__ __forceinline__ v16bf load_afrag(const unsigned short* p) {
  BFrag f;
  f.q[0] = *(const uint4*)(p);
  f.q[1] = *(const uint4*)(p + 16);
  return f.v;
}

// ---------------- 1) weight packing: fp32 -> bf16, WMMA-B-fragment order ----------------
__global__ void pack_all(const float* __restrict__ wg, const float* __restrict__ wu,
                         const float* __restrict__ wd, const float* __restrict__ sg,
                         const float* __restrict__ su, const float* __restrict__ sd,
                         unsigned short* __restrict__ G, unsigned short* __restrict__ U,
                         unsigned short* __restrict__ D) {
  const int mat = blockIdx.y;            // 0..26
  const int which = mat / NEXP;          // 0=gate 1=up 2=down
  const int e = mat % NEXP;
  const float* src;
  unsigned short* dst;
  int Kd, Nd;
  if (which == 0)      { src = (e < Ee) ? wg + (size_t)e * Hh * Ii : sg; dst = G + (size_t)e * Hh * Ii; Kd = Hh; Nd = Ii; }
  else if (which == 1) { src = (e < Ee) ? wu + (size_t)e * Hh * Ii : su; dst = U + (size_t)e * Hh * Ii; Kd = Hh; Nd = Ii; }
  else                 { src = (e < Ee) ? wd + (size_t)e * Ii * Hh : sd; dst = D + (size_t)e * Ii * Hh; Kd = Ii; Nd = Hh; }

  const int idx = blockIdx.x * 256 + threadIdx.x;     // < Kd*Nd == 524288
  const int kk   = idx & 15;
  const int lane = (idx >> 4) & 31;
  const int rest = idx >> 9;
  const int KT = Kd >> 5;
  const int kt = rest % KT;
  const int nt = rest / KT;
  const int n = nt * 16 + (lane & 15);
  const int k = kt * 32 + (lane >> 4) * 8 + (kk & 7) + (kk >> 3) * 16;
  dst[idx] = f2bf(src[(size_t)k * Nd + n]);
}

// ---------------- 2) x fp32 -> bf16 ----------------
__global__ void cvt_x(const float* __restrict__ x, unsigned short* __restrict__ xbf) {
  const int i = blockIdx.x * 256 + threadIdx.x;
  xbf[i] = f2bf(x[i]);
}

// ---------------- 3) router: one wave32 per token ----------------
__global__ void router(const float* __restrict__ x, const float* __restrict__ rw,
                       const float* __restrict__ bias,
                       int* __restrict__ topk_idx, float* __restrict__ topk_w) {
  const int tok  = (blockIdx.x * blockDim.x + threadIdx.x) >> 5;
  const int lane = threadIdx.x & 31;
  if (tok >= Tt) return;
  const float* xr = x + (size_t)tok * Hh;

  float acc[Ee] = {};
  for (int h = lane; h < Hh; h += 32) {
    const float xv = xr[h];
#pragma unroll
    for (int e = 0; e < Ee; ++e) acc[e] += xv * rw[e * Hh + h];
  }
#pragma unroll
  for (int e = 0; e < Ee; ++e)
    for (int off = 16; off; off >>= 1) acc[e] += __shfl_xor(acc[e], off, 32);

  if (lane == 0) {
    float s[Ee], sc[Ee];
#pragma unroll
    for (int e = 0; e < Ee; ++e) {
      s[e]  = fast_sigmoid(acc[e]);
      sc[e] = s[e] + bias[e];
    }
    float gs[2];
#pragma unroll
    for (int g = 0; g < 2; ++g) {
      const int b = g * 4;
      float m1 = -1e30f, m2 = -1e30f;
#pragma unroll
      for (int j = 0; j < 4; ++j) {
        const float v = sc[b + j];
        if (v > m1) { m2 = m1; m1 = v; } else if (v > m2) { m2 = v; }
      }
      gs[g] = m1 + m2;
    }
    const int gsel = (gs[1] > gs[0]) ? 1 : 0;
    const int base = gsel * 4;
    int i1 = base;
#pragma unroll
    for (int j = 1; j < 4; ++j) if (sc[base + j] > sc[i1]) i1 = base + j;
    int i2 = -1; float b2 = -1e30f;
#pragma unroll
    for (int j = 0; j < 4; ++j) {
      const int e = base + j;
      if (e != i1 && sc[e] > b2) { b2 = sc[e]; i2 = e; }
    }
    const float w1 = s[i1], w2 = s[i2];
    const float inv = ROUTED_SCALE / (w1 + w2 + 1e-20f);
    topk_idx[2 * tok + 0] = i1;  topk_w[2 * tok + 0] = w1 * inv;
    topk_idx[2 * tok + 1] = i2;  topk_w[2 * tok + 1] = w2 * inv;
  }
}

// ---------------- 4) per-expert token lists (shared = expert Ee, all tokens, w=1) ----------------
__global__ void build_lists(const int* __restrict__ topk_idx, const float* __restrict__ topk_w,
                            int* __restrict__ cnt, int* __restrict__ toks, float* __restrict__ wts) {
  const int t = blockIdx.x * blockDim.x + threadIdx.x;
  if (t >= Tt) return;
#pragma unroll
  for (int k = 0; k < 2; ++k) {
    const int e = topk_idx[2 * t + k];
    const float w = topk_w[2 * t + k];
    const int pos = atomicAdd(&cnt[e], 1);
    toks[(size_t)e * Tt + pos] = t;
    wts[(size_t)e * Tt + pos] = w;
  }
  toks[(size_t)Ee * Tt + t] = t;
  wts[(size_t)Ee * Tt + t] = 1.0f;
  if (t == 0) cnt[Ee] = Tt;
}

// ---------------- 5) fused expert MLP, M-tile = 32 tokens ----------------
// Each loaded B fragment feeds 2 WMMAs (two 16-row M-subtiles): halves L2 weight traffic.
__global__ __launch_bounds__(256) void moe_gemm(
    const unsigned short* __restrict__ xbf, const int* __restrict__ cnt,
    const int* __restrict__ toks, const float* __restrict__ wts,
    const unsigned short* __restrict__ G, const unsigned short* __restrict__ U,
    const unsigned short* __restrict__ D, float* __restrict__ out) {
  const int e = blockIdx.y;
  const int tile = blockIdx.x;
  const int ne = cnt[e];
  if (tile * MT >= ne) return;

  __shared__ __align__(16) unsigned short xA[MT * Hh];   // 64 KB; reused as hA after stage 1
  unsigned short* hA = xA;                               // 32 rows x 512 bf16 = 32 KB

  const int tid = threadIdx.x;
  const int* elist = toks + (size_t)e * Tt;
  const float* ewts = wts + (size_t)e * Tt;

  { // gather 32 token rows (bf16) into LDS, 128B per thread
    const int m = tid >> 3, c = tid & 7;
    const int i = tile * MT + m;
    const int tok = (i < ne) ? elist[i] : 0;
    const uint4* s4 = (const uint4*)(xbf + (size_t)tok * Hh + c * 128);
    uint4* d4 = (uint4*)(&xA[m * Hh + c * 128]);
#pragma unroll
    for (int j = 0; j < 8; ++j) d4[j] = s4[j];
  }
  __syncthreads();

  const int wave = tid >> 5, lane = tid & 31;
  const int m = lane & 15, kb = (lane >> 4) * 8;

  // ---- stage 1: h = silu(x@Wg) * (x@Wu); wave owns cols [wave*64, wave*64+64)
  const size_t mgu = (size_t)e * Hh * Ii;
  constexpr int KT1 = Hh / 32;   // 32
  v8f accg[2][4] = {}, accu[2][4] = {};
  for (int kt = 0; kt < KT1; ++kt) {
    const v16bf a0 = load_afrag(&xA[m * Hh + kt * 32 + kb]);
    const v16bf a1 = load_afrag(&xA[(16 + m) * Hh + kt * 32 + kb]);
#pragma unroll
    for (int j = 0; j < 4; ++j) {
      const int nt = wave * 4 + j;
      const size_t fo = mgu + ((size_t)(nt * KT1 + kt) * 32 + lane) * 16;
      const v16bf bG = load_bfrag(G + fo);
      const v16bf bU = load_bfrag(U + fo);
      accg[0][j] = __builtin_amdgcn_wmma_f32_16x16x32_bf16(false, a0, false, bG, (short)0, accg[0][j], false, false);
      accg[1][j] = __builtin_amdgcn_wmma_f32_16x16x32_bf16(false, a1, false, bG, (short)0, accg[1][j], false, false);
      accu[0][j] = __builtin_amdgcn_wmma_f32_16x16x32_bf16(false, a0, false, bU, (short)0, accu[0][j], false, false);
      accu[1][j] = __builtin_amdgcn_wmma_f32_16x16x32_bf16(false, a1, false, bU, (short)0, accu[1][j], false, false);
    }
  }
  __syncthreads();   // all waves done reading xA; safe to alias with hA

  // SiLU(g)*u -> bf16 intermediate (D layout: M = r + 8*(lane>>4), N = lane&15)
#pragma unroll
  for (int mt = 0; mt < 2; ++mt) {
#pragma unroll
    for (int j = 0; j < 4; ++j) {
      const int ncol = wave * 64 + j * 16 + (lane & 15);
#pragma unroll
      for (int r = 0; r < 8; ++r) {
        const float g = accg[mt][j][r], u = accu[mt][j][r];
        const float h = g * fast_sigmoid(g) * u;
        hA[(mt * 16 + r + 8 * (lane >> 4)) * Ii + ncol] = f2bf(h);
      }
    }
  }
  __syncthreads();

  // ---- stage 2: y = h @ Wd ; wave owns cols [wave*128, wave*128+128)
  const size_t md = (size_t)e * Ii * Hh;
  constexpr int KT2 = Ii / 32;   // 16
  v8f acc2[2][8] = {};
  for (int kt = 0; kt < KT2; ++kt) {
    const v16bf a0 = load_afrag(&hA[m * Ii + kt * 32 + kb]);
    const v16bf a1 = load_afrag(&hA[(16 + m) * Ii + kt * 32 + kb]);
#pragma unroll
    for (int j = 0; j < 8; ++j) {
      const int nt = wave * 8 + j;
      const v16bf b = load_bfrag(D + md + ((size_t)(nt * KT2 + kt) * 32 + lane) * 16);
      acc2[0][j] = __builtin_amdgcn_wmma_f32_16x16x32_bf16(false, a0, false, b, (short)0, acc2[0][j], false, false);
      acc2[1][j] = __builtin_amdgcn_wmma_f32_16x16x32_bf16(false, a1, false, b, (short)0, acc2[1][j], false, false);
    }
  }

  // scale by routing weight, combine into output (skip tail rows entirely)
#pragma unroll
  for (int mt = 0; mt < 2; ++mt) {
#pragma unroll
    for (int r = 0; r < 8; ++r) {
      const int mr = mt * 16 + r + 8 * (lane >> 4);
      const int idx = tile * MT + mr;
      if (idx < ne) {
        const int tok = elist[idx];
        const float w = ewts[idx];
#pragma unroll
        for (int j = 0; j < 8; ++j) {
          const int ncol = wave * 128 + j * 16 + (lane & 15);
          atomicAdd(&out[(size_t)tok * Hh + ncol], acc2[mt][j][r] * w);
        }
      }
    }
  }
}

// ---------------- launch ----------------
extern "C" void kernel_launch(void* const* d_in, const int* in_sizes, int n_in,
                              void* d_out, int out_size, void* d_ws, size_t ws_size,
                              hipStream_t stream) {
  const float* x  = (const float*)d_in[0];
  const float* rw = (const float*)d_in[1];
  const float* cb = (const float*)d_in[2];
  const float* wg = (const float*)d_in[3];
  const float* wu = (const float*)d_in[4];
  const float* wd = (const float*)d_in[5];
  const float* sg = (const float*)d_in[6];
  const float* su = (const float*)d_in[7];
  const float* sd = (const float*)d_in[8];
  float* out = (float*)d_out;

  char* w = (char*)d_ws;
  size_t off = 0;
  auto alloc = [&](size_t bytes) -> void* {
    void* p = w + off;
    off += (bytes + 255) & ~(size_t)255;
    return p;
  };
  unsigned short* xbf = (unsigned short*)alloc((size_t)Tt * Hh * 2);
  int*   tidx = (int*)  alloc((size_t)Tt * 2 * 4);
  float* tw   = (float*)alloc((size_t)Tt * 2 * 4);
  int*   cnt  = (int*)  alloc(NEXP * 4);
  int*   toks = (int*)  alloc((size_t)NEXP * Tt * 4);
  float* wts  = (float*)alloc((size_t)NEXP * Tt * 4);
  unsigned short* G = (unsigned short*)alloc((size_t)NEXP * Hh * Ii * 2);
  unsigned short* U = (unsigned short*)alloc((size_t)NEXP * Hh * Ii * 2);
  unsigned short* D = (unsigned short*)alloc((size_t)NEXP * Ii * Hh * 2);
  (void)ws_size; (void)n_in; (void)in_sizes;

  hipMemsetAsync(cnt, 0, NEXP * 4, stream);
  hipMemsetAsync(out, 0, (size_t)out_size * sizeof(float), stream);

  pack_all<<<dim3((Hh * Ii) / 256, 3 * NEXP), 256, 0, stream>>>(wg, wu, wd, sg, su, sd, G, U, D);
  cvt_x<<<(Tt * Hh) / 256, 256, 0, stream>>>(x, xbf);
  router<<<Tt / 8, 256, 0, stream>>>(x, rw, cb, tidx, tw);
  build_lists<<<Tt / 256, 256, 0, stream>>>(tidx, tw, cnt, toks, wts);
  moe_gemm<<<dim3(Tt / MT, NEXP), 256, 0, stream>>>(xbf, cnt, toks, wts, G, U, D, out);
}
